// TransformerBranch_57878979281300
// MI455X (gfx1250) — matrix operand order repaired and checked by reading, not compiled, not run on hardware
//
#include <hip/hip_runtime.h>
#include <math.h>

typedef __bf16 bf16;
typedef __attribute__((ext_vector_type(16))) bf16 v16bf;
typedef __attribute__((ext_vector_type(8)))  bf16 bf8v;
typedef __attribute__((ext_vector_type(8)))  float v8f;

static constexpr int Bn = 8, Sn = 512, Dn = 768, NH = 8, DH = 96, Lc = 6, FF = 3072, PP = 1024;

__device__ inline bf16 f2bf(float f) {
  union { float f; unsigned u; } a; a.f = f;
  unsigned u = a.u;
  u += 0x7fffu + ((u >> 16) & 1u);           // round-to-nearest-even
  union { unsigned short s; bf16 b; } r; r.s = (unsigned short)(u >> 16);
  return r.b;
}

__device__ inline v8f vzero8() { v8f z = {0.f,0.f,0.f,0.f,0.f,0.f,0.f,0.f}; return z; }

__device__ inline bf8v bzero8() {
  union { unsigned short u[8]; bf8v v; } z = {{0,0,0,0,0,0,0,0}};
  return z.v;
}

// ---------------------------------------------------------------------------
// Batched bf16 GEMM, B pre-transposed: C[M,N] = A[M,K] * Bt[N,K]^T (+bias[N]).
// 128x128 tile / block, 256 threads (8 waves), wave tile 32x64 (8 WMMAs/k-step).
// Full tiles: double-buffered GLOBAL_LOAD_ASYNC_TO_LDS_B128 pipeline (ASYNCcnt);
// partial tiles: guarded single-buffer path.
// Epilogue MODE:
//  0: f32 row-major [m*ldc+n] (+z*strC)
//  1: bf16 head-major   [(b*NH+h)*sdim + s][dhdim] from m=(b,s), n=(h,d)
//  2: bf16 head-major^T [(b*NH+h)*dhdim + d][sdim] (V^T)
//  3: bf16 [b, s, NH*dhdim] with z=(b*NH+h): ctx merge
//  4: bf16 row-major with exact-erf GELU applied
// ---------------------------------------------------------------------------
template<int MODE>
__global__ __launch_bounds__(256) void gemm_kernel(
    const bf16* __restrict__ A, const bf16* __restrict__ Bt, const float* __restrict__ bias,
    void* __restrict__ C, int M, int N, int K, int lda, int ldbt, int ldc,
    long long strA, long long strB, long long strC, int sdim, int dhdim)
{
  __shared__ bf16 sA[2][128 * 32];   // [buf][row][k]
  __shared__ bf16 sB[2][128 * 32];   // [buf][n][k]
  const int z = blockIdx.z;
  A  += (long long)z * strA;
  Bt += (long long)z * strB;
  const int n0 = blockIdx.x * 128, m0 = blockIdx.y * 128;
  const int tid = threadIdx.x, lane = tid & 31, w = tid >> 5;
  const int half = lane >> 4, l16 = lane & 15;
  const int wm = (w & 3) * 32, wn = (w >> 2) * 64;   // wave tile origin in block tile
  const bool full = (m0 + 128 <= M) && (n0 + 128 <= N);

  v8f acc[2][4];
#pragma unroll
  for (int mi = 0; mi < 2; mi++)
#pragma unroll
    for (int nt = 0; nt < 4; nt++) acc[mi][nt] = vzero8();

  union F { v16bf v; bf8v h[2]; };

  auto compute = [&](int buf) {
    F afr[2];
#pragma unroll
    for (int mi = 0; mi < 2; mi++) {
      const bf16* ar = &sA[buf][(wm + mi * 16 + l16) * 32];
      afr[mi].h[0] = *(const bf8v*)(ar + half * 8);
      afr[mi].h[1] = *(const bf8v*)(ar + 16 + half * 8);
    }
#pragma unroll
    for (int nt = 0; nt < 4; nt++) {
      F bfr;
      const bf16* br = &sB[buf][(wn + nt * 16 + l16) * 32];
      bfr.h[0] = *(const bf8v*)(br + half * 8);
      bfr.h[1] = *(const bf8v*)(br + 16 + half * 8);
#pragma unroll
      for (int mi = 0; mi < 2; mi++)
        acc[mi][nt] = __builtin_amdgcn_wmma_f32_16x16x32_bf16(false, afr[mi].v, false, bfr.v,
                                                              (short)0, acc[mi][nt], false, false);
    }
  };

  if (full) {
    auto stage_async = [&](int k0, int buf) {
#pragma unroll
      for (int i = 0; i < 2; i++) {
        int f = tid * 8 + i * 2048;
        int row = f >> 5, col = f & 31;
        unsigned long long ga = (unsigned long long)(A + (long long)(m0 + row) * lda + (k0 + col));
        unsigned la = (unsigned)(unsigned long long)(&sA[buf][f]);
        asm volatile("global_load_async_to_lds_b128 %0, %1, off" :: "v"(la), "v"(ga) : "memory");
        unsigned long long gb = (unsigned long long)(Bt + (long long)(n0 + row) * ldbt + (k0 + col));
        unsigned lb = (unsigned)(unsigned long long)(&sB[buf][f]);
        asm volatile("global_load_async_to_lds_b128 %0, %1, off" :: "v"(lb), "v"(gb) : "memory");
      }
    };
    stage_async(0, 0);
    asm volatile("s_wait_asynccnt 0x0" ::: "memory");
    __syncthreads();
    int buf = 0;
    for (int k0 = 0; k0 < K; k0 += 32) {
      if (k0 + 32 < K) stage_async(k0 + 32, buf ^ 1);   // prefetch next tile (async DMA)
      compute(buf);                                     // WMMAs overlap the DMA
      if (k0 + 32 < K) {
        asm volatile("s_wait_asynccnt 0x0" ::: "memory");
        __syncthreads();
      }
      buf ^= 1;
    }
  } else {
    for (int k0 = 0; k0 < K; k0 += 32) {
#pragma unroll
      for (int i = 0; i < 2; i++) {
        int f = tid * 8 + i * 2048;
        int row = f >> 5, col = f & 31;
        bf8v va = bzero8();
        if (m0 + row < M) va = *(const bf8v*)(A + (long long)(m0 + row) * lda + (k0 + col));
        *(bf8v*)(&sA[0][f]) = va;
        bf8v vb = bzero8();
        if (n0 + row < N) vb = *(const bf8v*)(Bt + (long long)(n0 + row) * ldbt + (k0 + col));
        *(bf8v*)(&sB[0][f]) = vb;
      }
      __syncthreads();
      compute(0);
      __syncthreads();
    }
  }

  // ---- epilogue (templated, slim) ----
#pragma unroll
  for (int mi = 0; mi < 2; mi++) {
    const int rowb = m0 + wm + mi * 16 + half * 8;
    int b_ = 0, s0 = 0, hz = 0;
    if constexpr (MODE == 1 || MODE == 2) { b_ = rowb / sdim; s0 = rowb - b_ * sdim; }
    if constexpr (MODE == 3) { b_ = z / NH; hz = z - b_ * NH; }
#pragma unroll
    for (int nt = 0; nt < 4; nt++) {
      int n = n0 + wn + nt * 16 + l16;
      if (n < N) {
        float bv = bias ? bias[n] : 0.f;
        int hh = 0, dd = n;
        if constexpr (MODE == 1 || MODE == 2) { hh = n / dhdim; dd = n - hh * dhdim; }
#pragma unroll
        for (int r = 0; r < 8; r++) {
          int m = rowb + r;
          if (m < M) {
            float v = acc[mi][nt][r] + bv;
            if constexpr (MODE == 0) {
              ((float*)C)[(long long)z * strC + (long long)m * ldc + n] = v;
            } else if constexpr (MODE == 1) {
              ((bf16*)C)[(((long long)(b_ * NH + hh) * sdim + (s0 + r)) * dhdim) + dd] = f2bf(v);
            } else if constexpr (MODE == 2) {
              ((bf16*)C)[(((long long)(b_ * NH + hh) * dhdim + dd) * sdim) + (s0 + r)] = f2bf(v);
            } else if constexpr (MODE == 3) {
              ((bf16*)C)[((long long)(b_ * sdim + m) * (NH * dhdim)) + hz * dhdim + n] = f2bf(v);
            } else {
              float gv = 0.5f * v * (1.f + erff(v * 0.7071067811865475f));
              ((bf16*)C)[(long long)m * ldc + n] = f2bf(gv);
            }
          }
        }
      }
    }
  }
}

// ---------------------------------------------------------------------------
// Fused DeBERTa disentangled attention scores.
// One wave per 16x16 (q,k) tile: QK^T + c2p + p2c, gathered via LDS slabs.
// scores(q,k) = [ q.k + q.pk[clip(q-k+512)] + k.pq[clip(q-k+512)] ] / sqrt(3*DH)
// ---------------------------------------------------------------------------
__global__ __launch_bounds__(128) void attn_scores_kernel(
    const bf16* __restrict__ qh, const bf16* __restrict__ kh,
    const bf16* __restrict__ pkh, const bf16* __restrict__ pqh,
    float* __restrict__ scores)
{
  __shared__ float lc2p[4][16][32];
  __shared__ float lp2c[4][16][32];
  const int tid = threadIdx.x, lane = tid & 31, w = tid >> 5;
  const int half = lane >> 4, l16 = lane & 15;
  const int kt = blockIdx.x * 4 + w;
  const int q0 = blockIdx.y * 16, k0 = kt * 16;
  const int bh = blockIdx.z, hh = bh & (NH - 1);
  const bf16* Q  = qh  + (long long)bh * Sn * DH;
  const bf16* Kt = kh  + (long long)bh * Sn * DH;
  const bf16* PK = pkh + (long long)hh * PP * DH;
  const bf16* PQ = pqh + (long long)hh * PP * DH;
  const int base = q0 - k0 + 512;
  int p0 = base + l16 - 16;       p0 = p0 < 0 ? 0 : (p0 > PP - 1 ? PP - 1 : p0);
  int p1 = base + l16;            p1 = p1 < 0 ? 0 : (p1 > PP - 1 ? PP - 1 : p1);

  v8f qk = vzero8(), c2p0 = vzero8(), c2p1 = vzero8(), p2c0 = vzero8(), p2c1 = vzero8();
  union F { v16bf v; bf8v h[2]; };
#pragma unroll
  for (int kk = 0; kk < DH; kk += 32) {
    const int kb = kk + half * 16;
    F aq, ak, bk, bp0, bp1, bq0, bq1;
    const bf16* qr = Q + (long long)(q0 + l16) * DH + kk;       // A frag: Q rows
    aq.h[0] = *(const bf8v*)(qr + half * 16);
    aq.h[1] = *(const bf8v*)(qr + half * 16 + 8);
    const bf16* krw = Kt + (long long)(k0 + l16) * DH + kk;     // A frag: K rows
    ak.h[0] = *(const bf8v*)(krw + half * 16);
    ak.h[1] = *(const bf8v*)(krw + half * 16 + 8);
    const bf16* kc = Kt + (long long)(k0 + l16) * DH + kb;      // B frag: K^T cols = keys
    bk.h[0] = *(const bf8v*)(kc);      bk.h[1] = *(const bf8v*)(kc + 8);
    const bf16* a0 = PK + (long long)p0 * DH + kb;              // B frag: pk^T slab cols
    bp0.h[0] = *(const bf8v*)(a0);     bp0.h[1] = *(const bf8v*)(a0 + 8);
    const bf16* a1 = PK + (long long)p1 * DH + kb;
    bp1.h[0] = *(const bf8v*)(a1);     bp1.h[1] = *(const bf8v*)(a1 + 8);
    const bf16* b0 = PQ + (long long)p0 * DH + kb;              // B frag: pq^T slab cols
    bq0.h[0] = *(const bf8v*)(b0);     bq0.h[1] = *(const bf8v*)(b0 + 8);
    const bf16* b1 = PQ + (long long)p1 * DH + kb;
    bq1.h[0] = *(const bf8v*)(b1);     bq1.h[1] = *(const bf8v*)(b1 + 8);

    qk   = __builtin_amdgcn_wmma_f32_16x16x32_bf16(false, aq.v, false, bk.v,  (short)0, qk,   false, false);
    c2p0 = __builtin_amdgcn_wmma_f32_16x16x32_bf16(false, aq.v, false, bp0.v, (short)0, c2p0, false, false);
    c2p1 = __builtin_amdgcn_wmma_f32_16x16x32_bf16(false, aq.v, false, bp1.v, (short)0, c2p1, false, false);
    p2c0 = __builtin_amdgcn_wmma_f32_16x16x32_bf16(false, ak.v, false, bq0.v, (short)0, p2c0, false, false);
    p2c1 = __builtin_amdgcn_wmma_f32_16x16x32_bf16(false, ak.v, false, bq1.v, (short)0, p2c1, false, false);
  }

  // dump slabs (C layout: lane%16 = col, VGPR r -> row 8*(lane/16)+r)
#pragma unroll
  for (int r = 0; r < 8; r++) {
    int m = half * 8 + r;
    lc2p[w][m][l16]      = c2p0[r];
    lc2p[w][m][16 + l16] = c2p1[r];
    lp2c[w][m][l16]      = p2c0[r];
    lp2c[w][m][16 + l16] = p2c1[r];
  }
  __syncthreads();

  const float invs = 0.05892556509887896f;  // 1/sqrt(3*96)
  float* out = scores + ((long long)bh * Sn + q0) * Sn + k0;
  const int dk = l16;
#pragma unroll
  for (int r = 0; r < 8; r++) {
    int dq = half * 8 + r;
    int j = dq - dk + 16;                   // in [1,31]
    float v = qk[r] + lc2p[w][dq][j] + lp2c[w][dk][j];
    out[(long long)dq * Sn + dk] = v * invs;
  }
}

// ---------------------------------------------------------------------------
__global__ __launch_bounds__(128) void softmax_kernel(
    const float* __restrict__ scores, const int* __restrict__ amask, bf16* __restrict__ probs)
{
  const int row = blockIdx.x;               // bh*S + q
  const int q = row & (Sn - 1);
  const int bh = row >> 9;
  const int b = bh >> 3;
  const float* src = scores + (long long)row * Sn;
  bf16* dst = probs + (long long)row * Sn;
  const int t = threadIdx.x;
  const int mq = amask[b * Sn + q];
  const float NEG = -3.4028234663852886e38f;
  float vals[4]; float mx = NEG;
#pragma unroll
  for (int i = 0; i < 4; i++) {
    int c = t + i * 128;
    int mk = amask[b * Sn + c];
    float x = (mq != 0 && mk != 0) ? src[c] : NEG;
    vals[i] = x; mx = fmaxf(mx, x);
  }
  __shared__ float red[128];
  red[t] = mx; __syncthreads();
  for (int s = 64; s > 0; s >>= 1) { if (t < s) red[t] = fmaxf(red[t], red[t + s]); __syncthreads(); }
  mx = red[0]; __syncthreads();
  float sum = 0.f;
#pragma unroll
  for (int i = 0; i < 4; i++) { vals[i] = __expf(vals[i] - mx); sum += vals[i]; }
  red[t] = sum; __syncthreads();
  for (int s = 64; s > 0; s >>= 1) { if (t < s) red[t] += red[t + s]; __syncthreads(); }
  float inv = 1.f / red[0];
#pragma unroll
  for (int i = 0; i < 4; i++) { int c = t + i * 128; dst[c] = f2bf(vals[i] * inv); }
}

// ---------------------------------------------------------------------------
__global__ __launch_bounds__(256) void embed_ln_kernel(
    const float* __restrict__ x, const float* __restrict__ pos,
    const float* __restrict__ g, const float* __restrict__ bta,
    const int* __restrict__ amask, float* __restrict__ hout, bf16* __restrict__ hbf)
{
  const int row = blockIdx.x;               // b*S + s
  const int s = row & (Sn - 1);
  const int t = threadIdx.x;
  const float* xr = x + (long long)row * Dn;
  const float* pr = pos + (long long)s * Dn;
  float v[3]; float lsum = 0.f;
#pragma unroll
  for (int i = 0; i < 3; i++) { int c = t + i * 256; v[i] = xr[c] + pr[c]; lsum += v[i]; }
  __shared__ float red[256];
  red[t] = lsum; __syncthreads();
  for (int ss = 128; ss > 0; ss >>= 1) { if (t < ss) red[t] += red[t + ss]; __syncthreads(); }
  float mean = red[0] * (1.f / Dn); __syncthreads();
  float lv = 0.f;
#pragma unroll
  for (int i = 0; i < 3; i++) { float d = v[i] - mean; lv += d * d; }
  red[t] = lv; __syncthreads();
  for (int ss = 128; ss > 0; ss >>= 1) { if (t < ss) red[t] += red[t + ss]; __syncthreads(); }
  float inv = rsqrtf(red[0] * (1.f / Dn) + 1e-7f);
  float mk = (float)amask[row];
#pragma unroll
  for (int i = 0; i < 3; i++) {
    int c = t + i * 256;
    float o = ((v[i] - mean) * inv * g[c] + bta[c]) * mk;
    hout[(long long)row * Dn + c] = o;
    hbf [(long long)row * Dn + c] = f2bf(o);
  }
}

__global__ __launch_bounds__(256) void ln_res_kernel(
    const float* __restrict__ a, const float* __restrict__ res,
    const float* __restrict__ g, const float* __restrict__ bta,
    float* __restrict__ hout, bf16* __restrict__ hbf)
{
  const int row = blockIdx.x;
  const int t = threadIdx.x;
  const float* ar = a + (long long)row * Dn;
  const float* rr = res + (long long)row * Dn;
  float v[3]; float lsum = 0.f;
#pragma unroll
  for (int i = 0; i < 3; i++) { int c = t + i * 256; v[i] = ar[c] + rr[c]; lsum += v[i]; }
  __shared__ float red[256];
  red[t] = lsum; __syncthreads();
  for (int ss = 128; ss > 0; ss >>= 1) { if (t < ss) red[t] += red[t + ss]; __syncthreads(); }
  float mean = red[0] * (1.f / Dn); __syncthreads();
  float lv = 0.f;
#pragma unroll
  for (int i = 0; i < 3; i++) { float d = v[i] - mean; lv += d * d; }
  red[t] = lv; __syncthreads();
  for (int ss = 128; ss > 0; ss >>= 1) { if (t < ss) red[t] += red[t + ss]; __syncthreads(); }
  float inv = rsqrtf(red[0] * (1.f / Dn) + 1e-7f);
#pragma unroll
  for (int i = 0; i < 3; i++) {
    int c = t + i * 256;
    float o = (v[i] - mean) * inv * g[c] + bta[c];
    hout[(long long)row * Dn + c] = o;
    hbf[(long long)row * Dn + c] = f2bf(o);
  }
}

__global__ void cvt_bf_kernel(const float* __restrict__ a, bf16* __restrict__ o, long long n) {
  long long i = (long long)blockIdx.x * 256 + threadIdx.x;
  if (i < n) o[i] = f2bf(a[i]);
}

// Tiled transpose+cast: in f32 [z][K][N] -> out bf16 [z][N][K]
__global__ __launch_bounds__(256) void transpose_cvt_kernel(
    const float* __restrict__ in, bf16* __restrict__ out, int K, int N)
{
  __shared__ float tile[32][33];
  const long long zoff = (long long)blockIdx.z * K * N;
  in += zoff; out += zoff;
  const int k0 = blockIdx.y * 32, n0 = blockIdx.x * 32;
  const int tx = threadIdx.x & 31, ty = threadIdx.x >> 5;   // 32 x 8
#pragma unroll
  for (int i = 0; i < 32; i += 8) {
    int k = k0 + ty + i, n = n0 + tx;
    tile[ty + i][tx] = (k < K && n < N) ? in[(long long)k * N + n] : 0.f;
  }
  __syncthreads();
#pragma unroll
  for (int i = 0; i < 32; i += 8) {
    int n = n0 + ty + i, k = k0 + tx;
    if (n < N && k < K) out[(long long)n * K + k] = f2bf(tile[tx][ty + i]);
  }
}

// ---------------------------------------------------------------------------
extern "C" void kernel_launch(void* const* d_in, const int* in_sizes, int n_in,
                              void* d_out, int out_size, void* d_ws, size_t ws_size,
                              hipStream_t stream)
{
  const float* x      = (const float*)d_in[0];
  const float* pos    = (const float*)d_in[1];
  const float* rel    = (const float*)d_in[2];
  const float* ln_e_g = (const float*)d_in[3];
  const float* ln_e_b = (const float*)d_in[4];
  const float* Wq  = (const float*)d_in[5];  const float* bq  = (const float*)d_in[6];
  const float* Wk  = (const float*)d_in[7];  const float* bk  = (const float*)d_in[8];
  const float* Wv  = (const float*)d_in[9];  const float* bv  = (const float*)d_in[10];
  const float* Wo  = (const float*)d_in[11]; const float* bo  = (const float*)d_in[12];
  const float* Wpk = (const float*)d_in[13]; const float* bpk = (const float*)d_in[14];
  const float* Wpq = (const float*)d_in[15]; const float* bpq = (const float*)d_in[16];
  const float* ln1g = (const float*)d_in[17]; const float* ln1b = (const float*)d_in[18];
  const float* Wi  = (const float*)d_in[19]; const float* bi  = (const float*)d_in[20];
  const float* Wo2 = (const float*)d_in[21]; const float* bo2 = (const float*)d_in[22];
  const float* ln2g = (const float*)d_in[23]; const float* ln2b = (const float*)d_in[24];
  const int*   amask = (const int*)d_in[25];
  (void)in_sizes; (void)n_in; (void)out_size; (void)ws_size;

  const long long DD  = (long long)Dn * Dn;
  const long long BS  = (long long)Bn * Sn;
  const long long BSD = BS * Dn;

  char* ws = (char*)d_ws;
  size_t off = 0;
  auto carve = [&](long long bytes) -> char* {
    off = (off + 255) & ~(size_t)255;
    char* p = ws + off; off += (size_t)bytes; return p;
  };
  bf16* WqT  = (bf16*)carve(Lc * DD * 2);
  bf16* WkT  = (bf16*)carve(Lc * DD * 2);
  bf16* WvT  = (bf16*)carve(Lc * DD * 2);
  bf16* WoT  = (bf16*)carve(Lc * DD * 2);
  bf16* WpkT = (bf16*)carve(Lc * DD * 2);
  bf16* WpqT = (bf16*)carve(Lc * DD * 2);
  bf16* WiT  = (bf16*)carve((long long)Lc * Dn * FF * 2);
  bf16* Wo2T = (bf16*)carve((long long)Lc * FF * Dn * 2);
  bf16* relB = (bf16*)carve((long long)PP * Dn * 2);
  float* h   = (float*)carve(BSD * 4);
  bf16*  hB  = (bf16*) carve(BSD * 2);
  float* tmp = (float*)carve(BSD * 4);
  bf16*  qhB = (bf16*) carve(BSD * 2);
  bf16*  khB = (bf16*) carve(BSD * 2);
  bf16*  vT  = (bf16*) carve(BSD * 2);                      // [B,NH,DH,S]
  bf16*  pkh = (bf16*) carve((long long)NH * PP * DH * 2);
  bf16*  pqh = (bf16*) carve((long long)NH * PP * DH * 2);
  float* sc  = (float*)carve((long long)Bn * NH * Sn * Sn * 4);
  bf16*  prb = (bf16*) carve((long long)Bn * NH * Sn * Sn * 2);
  bf16*  ctxB= (bf16*) carve(BSD * 2);
  bf16*  ffB = (bf16*) carve(BS * FF * 2);

  auto tpose = [&](const float* s, bf16* d, int K_, int N_, int L_) {
    dim3 g((unsigned)((N_ + 31) / 32), (unsigned)((K_ + 31) / 32), (unsigned)L_);
    transpose_cvt_kernel<<<g, 256, 0, stream>>>(s, d, K_, N_);
  };
  tpose(Wq,  WqT,  Dn, Dn, Lc);  tpose(Wk,  WkT,  Dn, Dn, Lc);
  tpose(Wv,  WvT,  Dn, Dn, Lc);  tpose(Wo,  WoT,  Dn, Dn, Lc);
  tpose(Wpk, WpkT, Dn, Dn, Lc);  tpose(Wpq, WpqT, Dn, Dn, Lc);
  tpose(Wi,  WiT,  Dn, FF, Lc);
  tpose(Wo2, Wo2T, FF, Dn, Lc);
  cvt_bf_kernel<<<dim3((unsigned)((PP * Dn + 255) / 256)), 256, 0, stream>>>(
      rel, relB, (long long)PP * Dn);

  embed_ln_kernel<<<dim3((unsigned)BS), 256, 0, stream>>>(x, pos, ln_e_g, ln_e_b, amask, h, hB);

  auto gemm = [&](int mode, const bf16* A, const bf16* Bt, const float* bias, void* C,
                  int M, int N, int K, int lda, int ldbt, int ldc,
                  long long sa, long long sb, long long scv, int batch,
                  int sdim, int dhdim) {
    dim3 g((unsigned)((N + 127) / 128), (unsigned)((M + 127) / 128), (unsigned)batch);
    switch (mode) {
      case 0: gemm_kernel<0><<<g, 256, 0, stream>>>(A, Bt, bias, C, M, N, K, lda, ldbt, ldc, sa, sb, scv, sdim, dhdim); break;
      case 1: gemm_kernel<1><<<g, 256, 0, stream>>>(A, Bt, bias, C, M, N, K, lda, ldbt, ldc, sa, sb, scv, sdim, dhdim); break;
      case 2: gemm_kernel<2><<<g, 256, 0, stream>>>(A, Bt, bias, C, M, N, K, lda, ldbt, ldc, sa, sb, scv, sdim, dhdim); break;
      case 3: gemm_kernel<3><<<g, 256, 0, stream>>>(A, Bt, bias, C, M, N, K, lda, ldbt, ldc, sa, sb, scv, sdim, dhdim); break;
      default: gemm_kernel<4><<<g, 256, 0, stream>>>(A, Bt, bias, C, M, N, K, lda, ldbt, ldc, sa, sb, scv, sdim, dhdim); break;
    }
  };

  for (int l = 0; l < Lc; l++) {
    // Q/K projections -> bf16 [B,NH,S,DH]; V -> bf16 [B,NH,DH,S]
    gemm(1, hB, WqT + l * DD, bq + (long long)l * Dn, qhB, (int)BS, Dn, Dn, Dn, Dn, Dn,
         0, 0, 0, 1, Sn, DH);
    gemm(1, hB, WkT + l * DD, bk + (long long)l * Dn, khB, (int)BS, Dn, Dn, Dn, Dn, Dn,
         0, 0, 0, 1, Sn, DH);
    gemm(2, hB, WvT + l * DD, bv + (long long)l * Dn, vT, (int)BS, Dn, Dn, Dn, Dn, Dn,
         0, 0, 0, 1, Sn, DH);
    // relative-position projections -> bf16 [NH, 2*SPAN, DH]
    gemm(1, relB, WpkT + l * DD, bpk + (long long)l * Dn, pkh, PP, Dn, Dn, Dn, Dn, Dn,
         0, 0, 0, 1, PP, DH);
    gemm(1, relB, WpqT + l * DD, bpq + (long long)l * Dn, pqh, PP, Dn, Dn, Dn, Dn, Dn,
         0, 0, 0, 1, PP, DH);
    // fused scores (QK^T + c2p + p2c gathers) and masked softmax -> bf16 probs
    attn_scores_kernel<<<dim3(Sn / 64, Sn / 16, Bn * NH), 128, 0, stream>>>(qhB, khB, pkh, pqh, sc);
    softmax_kernel<<<dim3((unsigned)(Bn * NH * Sn)), 128, 0, stream>>>(sc, amask, prb);
    // ctx = P @ V (V supplied transposed), written straight to bf16 [B,S,D]
    gemm(3, prb, vT, nullptr, ctxB, Sn, DH, Sn, Sn, Sn, DH,
         (long long)Sn * Sn, (long long)Sn * DH, 0, Bn * NH, Sn, DH);
    // attention output proj + residual LN
    gemm(0, ctxB, WoT + l * DD, bo + (long long)l * Dn, tmp, (int)BS, Dn, Dn, Dn, Dn, Dn,
         0, 0, 0, 1, 0, 0);
    ln_res_kernel<<<dim3((unsigned)BS), 256, 0, stream>>>(
        tmp, h, ln1g + (long long)l * Dn, ln1b + (long long)l * Dn, h, hB);
    // FFN: GEMM1 with fused exact-GELU -> bf16, GEMM2 -> f32, residual LN
    gemm(4, hB, WiT + (long long)l * Dn * FF, bi + (long long)l * FF, ffB,
         (int)BS, FF, Dn, Dn, Dn, FF, 0, 0, 0, 1, 0, 0);
    gemm(0, ffB, Wo2T + (long long)l * FF * Dn, bo2 + (long long)l * Dn, tmp,
         (int)BS, Dn, FF, FF, FF, Dn, 0, 0, 0, 1, 0, 0);
    float* dstf = (l == Lc - 1) ? (float*)d_out : h;
    ln_res_kernel<<<dim3((unsigned)BS), 256, 0, stream>>>(
        tmp, h, ln2g + (long long)l * Dn, ln2b + (long long)l * Dn, dstf, hB);
  }
}
